// PsiQRHAutoCalConsciousnessPipeline_81913616269387
// MI455X (gfx1250) — matrix operand order repaired and checked by reading, not compiled, not run on hardware
//
#include <hip/hip_runtime.h>
#include <math.h>

typedef __attribute__((ext_vector_type(16))) _Float16 v16h;
typedef __attribute__((ext_vector_type(8)))  float    v8f;
typedef __attribute__((ext_vector_type(4)))  unsigned int u32x4;
typedef __attribute__((ext_vector_type(8)))  int      i32x8;
typedef __attribute__((ext_vector_type(4)))  int      i32x4;

#define D_MODEL 256
#define NFEAT   1024      // 4*D
#define SEQ     2048
#define BATCH   4
#define ROWS    (BATCH*SEQ)   // 8192
#define EPSF    1e-10f
#define TWO_PI  6.283185307179586f

// ---------------------------------------------------------------------------
// TDM: issue a 2-D tensor_load_to_lds (tile_x x tile_y elems of 4B) from
// global (row stride = stride_elems) into LDS byte offset lds_off.
// D# encoding per CDNA5 ISA 8.3/8.4 (count=1, type=2, data_size=4B,
// groups 2/3 zeroed => 2-D tile). Huge tensor_dim* => no OOB clipping.
// ---------------------------------------------------------------------------
__device__ __forceinline__ void tdm_load_2d(unsigned lds_off, const void* gaddr,
                                            int tile_x, int tile_y,
                                            unsigned stride_elems)
{
    unsigned long long ga = (unsigned long long)(uintptr_t)gaddr;
    u32x4 g0 = {0u, 0u, 0u, 0u};
    g0[0] = 1u;                                        // count=1, user mode
    g0[1] = lds_off;                                   // lds_addr (bytes)
    g0[2] = (unsigned)(ga & 0xFFFFFFFFu);              // global_addr[31:0]
    g0[3] = (unsigned)((ga >> 32) & 0x01FFFFFFu) | 0x80000000u; // [56:32]|type=2

    const unsigned td0 = 0x40000000u, td1 = 0x40000000u;   // huge tensor dims
    i32x8 g1 = {0, 0, 0, 0, 0, 0, 0, 0};
    g1[0] = (int)(2u << 16);                           // data_size=2 (4 bytes)
    g1[1] = (int)((td0 & 0xFFFFu) << 16);              // tensor_dim0[15:0]
    g1[2] = (int)(((td0 >> 16) & 0xFFFFu) | ((td1 & 0xFFFFu) << 16));
    g1[3] = (int)(((td1 >> 16) & 0xFFFFu) | ((unsigned)tile_x << 16)); // tile_dim0
    g1[4] = (int)((unsigned)tile_y & 0xFFFFu);         // tile_dim1 (tile_dim2=0)
    g1[5] = (int)stride_elems;                         // tensor_dim0_stride[31:0]
    g1[6] = 0;                                         // stride hi + dim1_stride lo
    g1[7] = 0;

    i32x4 zg = {0, 0, 0, 0};
#if defined(__clang_major__) && (__clang_major__ >= 23)
    i32x8 z8 = {0, 0, 0, 0, 0, 0, 0, 0};
    __builtin_amdgcn_tensor_load_to_lds(g0, g1, zg, zg, z8, 0);
#else
    __builtin_amdgcn_tensor_load_to_lds(g0, g1, zg, zg, 0);
#endif
}

// ---------------------------------------------------------------------------
// Kernel 1: psi = emb_table[input_ids] @ W_proj   ([8192,256]x[256,1024])
// One wave per 16x16 output tile; v_wmma_f32_16x16x32_f16, f32 accumulate.
// ---------------------------------------------------------------------------
__global__ void k_embed_project(const int* __restrict__ ids,
                                const float* __restrict__ emb,
                                const float* __restrict__ Wp,
                                float* __restrict__ psi)
{
    const int lane   = threadIdx.x & 31;
    const int tile   = blockIdx.x * (blockDim.x >> 5) + (threadIdx.x >> 5);
    const int tilesN = NFEAT / 16;              // 64
    const int tM = tile / tilesN;
    const int tN = tile - tM * tilesN;

    const int rowA   = lane & 15;
    const int kbaseA = (lane >> 4) << 3;        // 0 or 8
    const int kbaseB = (lane >> 4) << 4;        // 0 or 16
    const int colN   = tN * 16 + (lane & 15);

    const int tok = ids[tM * 16 + rowA];
    const float* arow = emb + (long)tok * D_MODEL;

    v8f c = {};
    for (int kk = 0; kk < D_MODEL; kk += 32) {
        if (kk + 32 < D_MODEL)
            __builtin_prefetch(Wp + (long)(kk + 32 + kbaseB) * NFEAT + colN, 0, 1);
        v16h a, b;
#pragma unroll
        for (int e = 0; e < 16; ++e) {
            int ka = kk + ((e >= 8) ? 16 : 0) + kbaseA + (e & 7);
            a[e] = (_Float16)arow[ka];
            int kb = kk + ((e >= 8) ? 16 : 0) + kbaseA + (e & 7);
            b[e] = (_Float16)Wp[kb * NFEAT + colN];
        }
        c = __builtin_amdgcn_wmma_f32_16x16x32_f16(false, a, false, b,
                                                   (short)0, c, false, false);
    }
    const int rbase = tM * 16 + ((lane >> 4) << 3);
#pragma unroll
    for (int r = 0; r < 8; ++r)
        psi[(long)(rbase + r) * NFEAT + colN] = c[r];
}

// ---------------------------------------------------------------------------
// Kernel 2: ps[k] = mean_{b,s} |DFT_256(psi[b,s,:,0])[k]|^2
// ---------------------------------------------------------------------------
__global__ void k_power_spectrum(const float* __restrict__ psi,
                                 float* __restrict__ ps)
{
    __shared__ float x[8][D_MODEL];   // 8 KB
    __shared__ float tab[D_MODEL];    // 1 KB
    const int k    = threadIdx.x;     // 0..255
    const int row0 = blockIdx.x * 8;
    tab[k] = cosf(TWO_PI * (float)k / 256.0f);
#pragma unroll
    for (int r = 0; r < 8; ++r)
        x[r][k] = psi[(long)(row0 + r) * NFEAT + k * 4];   // component 0
    __syncthreads();
    float acc = 0.f;
#pragma unroll
    for (int r = 0; r < 8; ++r) {
        float re = 0.f, im = 0.f;
        for (int d = 0; d < D_MODEL; ++d) {
            int idx = (k * d) & 255;
            re += x[r][d] * tab[idx];
            im -= x[r][d] * tab[(idx + 192) & 255];   // sin t = cos(t-pi/2)
        }
        acc += re * re + im * im;
    }
    atomicAdd(&ps[k], acc * (1.0f / (float)ROWS));
}

// ---------------------------------------------------------------------------
// Kernel 3: alpha (log-log slope of ps[1..127]) + quaternion coeffs.
// ---------------------------------------------------------------------------
__global__ void k_alpha_quat(const float* __restrict__ ps,
                             const float* __restrict__ thetaL,
                             float* __restrict__ params)
{
    if (threadIdx.x != 0) return;
    float mlk = 0.f, mlp = 0.f;
    for (int k = 1; k < 128; ++k) {
        mlk += logf((float)k);
        mlp += logf(ps[k] + EPSF);
    }
    mlk *= (1.0f / 127.0f);  mlp *= (1.0f / 127.0f);
    float num = 0.f, den = 0.f;
    for (int k = 1; k < 128; ++k) {
        float lk = logf((float)k) - mlk;
        num += lk * (logf(ps[k] + EPSF) - mlp);
        den += lk * lk;
    }
    float alpha = -(num / den) * 0.5f;
    alpha = fminf(fmaxf(alpha, 0.5f), 2.0f);
    params[0] = alpha;
    float th = thetaL[0], om = thetaL[1], ph = thetaL[2];
    float s = sinf(th);
    params[1] = cosf(th);
    params[2] = s * cosf(om);
    params[3] = s * sinf(om) * cosf(ph);
    params[4] = s * sinf(om) * sinf(ph);
}

// ---------------------------------------------------------------------------
// Kernel 4: cphi[k] = cos(alpha * atan(log(|fftfreq(k)| + eps)))  (even in k)
// ---------------------------------------------------------------------------
__global__ void k_cphi(const float* __restrict__ params,
                       float* __restrict__ cphi)
{
    int k = blockIdx.x * blockDim.x + threadIdx.x;
    if (k >= SEQ) return;
    float alpha = params[0];
    int km = (k < SEQ - k) ? k : SEQ - k;
    float af = (float)km / (float)SEQ;
    cphi[k] = cosf(alpha * atanf(logf(af + EPSF)));
}

// ---------------------------------------------------------------------------
// Kernel 5: g[n] = (1/S) sum_k cphi[k] * cos(2*pi*k*n/S)
// ---------------------------------------------------------------------------
__global__ void k_build_g(const float* __restrict__ cphi,
                          float* __restrict__ g)
{
    __shared__ float tab[SEQ];   // 8 KB
    __shared__ float cf[SEQ];    // 8 KB
    for (int t = threadIdx.x; t < SEQ; t += blockDim.x) {
        tab[t] = cosf(TWO_PI * (float)t / (float)SEQ);
        cf[t]  = cphi[t];
    }
    __syncthreads();
    int n = blockIdx.x * blockDim.x + threadIdx.x;
    if (n >= SEQ) return;
    float acc = 0.f;
    for (int k = 0; k < SEQ; ++k)
        acc += cf[k] * tab[(k * n) & (SEQ - 1)];
    g[n] = acc * (1.0f / (float)SEQ);
}

// ---------------------------------------------------------------------------
// Kernel 6: attended_b = C @ psi_b, C[s,m] = g[(s-m) mod S]  (circulant GEMM)
// Block (8 waves) -> 128(M) x 64(N) tile.  TDM double-buffering:
//   wave 0 DMAs the next 32x64 f32 psi panel into the alternate LDS buffer
//   (tensor_load_to_lds, TENSORcnt) while all waves compute on the current
//   one; a cooperative pass converts f32 panel -> K-contiguous f16 pT so each
//   lane's B fragment is one contiguous 32B ds read.  g staged in LDS once.
// LDS map (manual): gt f16[2048] @0 | pbuf f32[2][32*64] @4096 | pT @20480.
// ---------------------------------------------------------------------------
#define SM_GT   0
#define SM_PB0  4096
#define SM_PB1  12288
#define SM_PT   20480
#define SM_SIZE 24576

__global__ void __launch_bounds__(256)
k_spectral_gemm(const float* __restrict__ g,
                const float* __restrict__ psi,
                float* __restrict__ att,
                const int* __restrict__ n_layers, int layer)
{
    if (layer >= *n_layers) return;

    __shared__ __align__(16) unsigned char smem[SM_SIZE];
    _Float16* gt = (_Float16*)(smem + SM_GT);             // 4 KB
    _Float16 (*pT)[32] = (_Float16(*)[32])(smem + SM_PT); // [64][32] f16

    const int tid  = threadIdx.x;
    const int lane = tid & 31;
    const int wave = tid >> 5;

    const int bid  = blockIdx.x;            // 0..1023
    const int b    = bid >> 8;
    const int rem  = bid & 255;
    const int tMb  = rem >> 4;              // 128-row strip
    const int tNb  = rem & 15;              // 64-col strip
    const int nbase = tNb * 64;

    const int colB   = lane & 15;
    const int kbaseA = (lane >> 4) << 3;    // 0 or 8
    const int kbaseB = (lane >> 4) << 4;    // 0 or 16
    const int sRow   = tMb * 128 + wave * 16 + (lane & 15);

    const float* xb = psi + (long)b * SEQ * NFEAT;

    for (int i = tid; i < SEQ; i += 256)    // stage g -> LDS (f16)
        gt[i] = (_Float16)g[i];

    // prime the pipeline: panel 0 -> pbuf[0]
    if (wave == 0)
        tdm_load_2d(SM_PB0, xb + nbase, 64, 32, NFEAT);

    v8f c0 = {}, c1 = {}, c2 = {}, c3 = {};

    for (int i = 0; i < SEQ / 32; ++i) {
        const int kk  = i * 32;
        const int cur = i & 1;
        if (wave == 0)
            __builtin_amdgcn_s_wait_tensorcnt(0);   // panel i landed in LDS
        __syncthreads();                            // panel visible; pT free

        if (wave == 0 && i + 1 < SEQ / 32)          // DMA next panel (async)
            tdm_load_2d(cur ? SM_PB0 : SM_PB1,
                        xb + (long)(kk + 32) * NFEAT + nbase, 64, 32, NFEAT);

        // convert f32 panel [k_local][n] -> f16 pT [n][k_local]
        // (pointer computed per-iteration; no LDS-pointer array initializer)
        const float* pcur = (const float*)(smem + (cur ? SM_PB1 : SM_PB0));
#pragma unroll
        for (int j = 0; j < 8; ++j) {
            int e  = tid * 8 + j;                   // 0..2047
            int ml = e >> 6;                        // k_local
            int nl = e & 63;                        // n_local
            pT[nl][ml] = (_Float16)pcur[ml * 64 + nl];
        }
        __syncthreads();                            // pT ready

        // A fragment from LDS g table (contiguous k = 16h + e, matching B)
        v16h a;
#pragma unroll
        for (int e = 0; e < 16; ++e) {
            int ma = kk + kbaseB + e;
            a[e] = gt[(sRow - ma) & (SEQ - 1)];
        }
        // 4 B fragments: contiguous 32B ds reads (K-major rows, 32B aligned)
        const v16h bf0 = *(const v16h*)&pT[ 0 + colB][kbaseB];
        const v16h bf1 = *(const v16h*)&pT[16 + colB][kbaseB];
        const v16h bf2 = *(const v16h*)&pT[32 + colB][kbaseB];
        const v16h bf3 = *(const v16h*)&pT[48 + colB][kbaseB];
        c0 = __builtin_amdgcn_wmma_f32_16x16x32_f16(false, a, false, bf0, (short)0, c0, false, false);
        c1 = __builtin_amdgcn_wmma_f32_16x16x32_f16(false, a, false, bf1, (short)0, c1, false, false);
        c2 = __builtin_amdgcn_wmma_f32_16x16x32_f16(false, a, false, bf2, (short)0, c2, false, false);
        c3 = __builtin_amdgcn_wmma_f32_16x16x32_f16(false, a, false, bf3, (short)0, c3, false, false);
    }

    float* ob = att + (long)b * SEQ * NFEAT;
    const int rbase = tMb * 128 + wave * 16 + ((lane >> 4) << 3);
#pragma unroll
    for (int r = 0; r < 8; ++r) {
        long ro = (long)(rbase + r) * NFEAT + nbase + colB;
        ob[ro +  0] = c0[r];
        ob[ro + 16] = c1[r];
        ob[ro + 32] = c2[r];
        ob[ro + 48] = c3[r];
    }
}

// ---------------------------------------------------------------------------
// Kernel 7: psi = normalize(quat_rotate(att) + psi); last layer also -> out.
// ---------------------------------------------------------------------------
__global__ void k_evolve(const float* __restrict__ att,
                         float* __restrict__ psi,
                         float* __restrict__ dout,
                         const float* __restrict__ params,
                         const int* __restrict__ n_layers, int layer)
{
    const int nl = *n_layers;
    long i = (long)blockIdx.x * blockDim.x + threadIdx.x;
    if (i >= (long)ROWS * D_MODEL) return;
    long base = i << 2;
    if (nl == 0) {                       // degenerate: output = psi0
        if (layer == 0) {
            dout[base]   = psi[base];   dout[base+1] = psi[base+1];
            dout[base+2] = psi[base+2]; dout[base+3] = psi[base+3];
        }
        return;
    }
    if (layer >= nl) return;
    float qw = params[1], qx = params[2], qy = params[3], qz = params[4];
    float aw = att[base], ax = att[base+1], ay = att[base+2], az = att[base+3];
    float pw = psi[base], px = psi[base+1], py = psi[base+2], pz = psi[base+3];
    float ew = qw*aw - qx*ax - qy*ay - qz*az;
    float ex = qw*ax + qx*aw + qy*az - qz*ay;
    float ey = qw*ay - qx*az + qy*aw + qz*ax;
    float ez = qw*az + qx*ay - qy*ax + qz*aw;
    float nw = ew + pw, nx = ex + px, ny = ey + py, nz = ez + pz;
    float inv = 1.0f / (sqrtf(nw*nw + nx*nx + ny*ny + nz*nz) + EPSF);
    nw *= inv; nx *= inv; ny *= inv; nz *= inv;
    psi[base] = nw; psi[base+1] = nx; psi[base+2] = ny; psi[base+3] = nz;
    if (layer == nl - 1) {
        dout[base] = nw; dout[base+1] = nx; dout[base+2] = ny; dout[base+3] = nz;
    }
}

// ---------------------------------------------------------------------------
extern "C" void kernel_launch(void* const* d_in, const int* in_sizes, int n_in,
                              void* d_out, int out_size, void* d_ws, size_t ws_size,
                              hipStream_t stream)
{
    const int*   ids = (const int*)d_in[0];
    const float* emb = (const float*)d_in[1];
    const float* Wp  = (const float*)d_in[2];
    const float* thL = (const float*)d_in[3];
    /* d_in[4] = theta_R : unused by the reference */
    const int*   nl  = (const int*)d_in[5];
    float* out = (float*)d_out;

    char* ws = (char*)d_ws;
    float* psi    = (float*)(ws);                 // 8192*1024 f32 = 32 MiB
    float* att    = (float*)(ws + 33554432);      // 32 MiB
    float* ps     = (float*)(ws + 67108864);      // 256 f32
    float* cphi   = (float*)(ws + 67109888);      // 2048 f32
    float* g      = (float*)(ws + 67118080);      // 2048 f32
    float* params = (float*)(ws + 67126272);      // 8 f32

    (void)hipMemsetAsync(ps, 0, 256 * sizeof(float), stream);

    // psi0 = gather+GEMM (WMMA): 512x64 tiles, 8 waves/block -> 4096 blocks
    k_embed_project<<<4096, 256, 0, stream>>>(ids, emb, Wp, psi);

    // alpha calibration + filter kernel build
    k_power_spectrum<<<ROWS / 8, 256, 0, stream>>>(psi, ps);
    k_alpha_quat<<<1, 32, 0, stream>>>(ps, thL, params);
    k_cphi<<<SEQ / 256, 256, 0, stream>>>(params, cphi);
    k_build_g<<<SEQ / 256, 256, 0, stream>>>(cphi, g);

    // layers (guarded in-kernel by device-side n_layers)
    for (int layer = 0; layer < 6; ++layer) {
        k_spectral_gemm<<<1024, 256, 0, stream>>>(g, psi, att, nl, layer);
        k_evolve<<<ROWS, 256, 0, stream>>>(att, psi, out, params, nl, layer);
    }
}
